// ONLSTM_77644418777253
// MI455X (gfx1250) — matrix operand order repaired and checked by reading, not compiled, not run on hardware
//
#include <hip/hip_runtime.h>
#include <hip/hip_bf16.h>
#include <math.h>

// ON-LSTM forward for MI455X (gfx1250), wave32 + WMMA bf16.
// B=64, T=512, D=U=1024, L=16, C=64, G=4*U+2*L=4128.

#define BDIM   64
#define TDIM   512
#define DDIM   1024
#define UDIM   1024
#define LVLS   16
#define GDIM   4128
#define NT_G   258     // GDIM / 16

typedef __attribute__((ext_vector_type(16))) __bf16 v16bf;
typedef __attribute__((ext_vector_type(8)))  float  v8f;

union BF16Frag { uint4 u[2]; v16bf v; };

static __device__ __forceinline__ unsigned short f32_to_bf16_rne(float f) {
  unsigned int u = __float_as_uint(f);
  u += 0x7FFFu + ((u >> 16) & 1u);   // round to nearest even
  return (unsigned short)(u >> 16);
}

// ---------------- prep kernels ----------------

__global__ void cvt_f32_to_bf16(const float* __restrict__ src,
                                unsigned short* __restrict__ dst, int n) {
  int i = blockIdx.x * blockDim.x + threadIdx.x;
  if (i < n) dst[i] = f32_to_bf16_rne(src[i]);
}

// wt[g*K + k] = bf16(w[k*G + g])  (coalesced read of w)
__global__ void transpose_to_bf16(const float* __restrict__ w,
                                  unsigned short* __restrict__ wt,
                                  int K, int G) {
  int i = blockIdx.x * blockDim.x + threadIdx.x;
  if (i >= K * G) return;
  int g = i % G;
  int k = i / G;
  wt[(size_t)g * K + k] = f32_to_bf16_rne(w[i]);
}

__global__ void init_state(float* __restrict__ c,
                           unsigned short* __restrict__ hb, int n) {
  int i = blockIdx.x * blockDim.x + threadIdx.x;
  if (i < n) { c[i] = 0.0f; hb[i] = 0; }
}

// ---------------- big GEMM: zx[t*B+b][g] = x[b,t,:] @ W + bias ----------------
// One wave computes a 16x32 output slab over K=1024 (64 WMMAs, A reused x2).

__global__ void __launch_bounds__(256) gemm_zx(
    const unsigned short* __restrict__ xb,   // bf16 x, original [B][T][D] layout
    const unsigned short* __restrict__ wt,   // bf16 W^T  [G][D]
    const float* __restrict__ bias,          // [G]
    float* __restrict__ zx)                  // f32 [T*B][G]
{
  const int waveId = (blockIdx.x * blockDim.x + threadIdx.x) >> 5;
  const int lane   = threadIdx.x & 31;
  const int Mtiles = (BDIM * TDIM) / 16;     // 2048
  const int Ngrp   = NT_G / 2;               // 129
  const int mt = waveId / Ngrp;
  const int ng = waveId % Ngrp;
  if (mt >= Mtiles) return;

  const int half = lane >> 4;
  const int mrow = mt * 16 + (lane & 15);    // global row r = t*B + b
  const int b    = mrow & (BDIM - 1);
  const int t    = mrow >> 6;
  const unsigned short* arow  = xb + ((size_t)b * TDIM + t) * DDIM;

  const int n0  = ng * 32;
  const int col = lane & 15;
  const unsigned short* brow0 = wt + (size_t)(n0 + col)      * DDIM;
  const unsigned short* brow1 = wt + (size_t)(n0 + 16 + col) * DDIM;

  v8f acc0 = {};
  v8f acc1 = {};
#pragma unroll 4
  for (int k0 = 0; k0 < DDIM; k0 += 32) {
    BF16Frag a;
    a.u[0] = *(const uint4*)(arow + k0 + 8 * half);
    a.u[1] = *(const uint4*)(arow + k0 + 16 + 8 * half);
    BF16Frag b0;
    b0.u[0] = *(const uint4*)(brow0 + k0 + 8 * half);
    b0.u[1] = *(const uint4*)(brow0 + k0 + 16 + 8 * half);
    BF16Frag b1;
    b1.u[0] = *(const uint4*)(brow1 + k0 + 8 * half);
    b1.u[1] = *(const uint4*)(brow1 + k0 + 16 + 8 * half);
    acc0 = __builtin_amdgcn_wmma_f32_16x16x32_bf16(false, a.v, false, b0.v,
                                                   (short)0, acc0, false, false);
    acc1 = __builtin_amdgcn_wmma_f32_16x16x32_bf16(false, a.v, false, b1.v,
                                                   (short)0, acc1, false, false);
  }

  const float bias0 = bias[n0 + col];
  const float bias1 = bias[n0 + 16 + col];
#pragma unroll
  for (int r = 0; r < 8; ++r) {
    const int mr = mt * 16 + r + 8 * half;
    const size_t zoff = (size_t)mr * GDIM;
    zx[zoff + n0 + col]      = acc0[r] + bias0;
    zx[zoff + n0 + 16 + col] = acc1[r] + bias1;
  }
}

// ---------------- per-step recurrent GEMM: z[b][g] += h @ R ----------------
// 4 M-tiles x 258 N-tiles = 1032 waves; Rt (8.25 MB) is L2-resident.

__global__ void __launch_bounds__(256) gemm_rec(
    const unsigned short* __restrict__ hb,   // bf16 [B][U]
    const unsigned short* __restrict__ rt,   // bf16 R^T [G][U]
    float* __restrict__ z)                   // f32 [B][G]  (zx slab for step t)
{
  const int waveId = (blockIdx.x * blockDim.x + threadIdx.x) >> 5;
  const int lane   = threadIdx.x & 31;
  const int mt = waveId / NT_G;
  const int nt = waveId % NT_G;
  if (mt >= BDIM / 16) return;

  const int half = lane >> 4;
  const int mrow = mt * 16 + (lane & 15);
  const int col  = lane & 15;
  const unsigned short* arow = hb + (size_t)mrow * UDIM;
  const unsigned short* brow = rt + (size_t)(nt * 16 + col) * UDIM;

  v8f acc = {};
#pragma unroll 4
  for (int k0 = 0; k0 < UDIM; k0 += 32) {
    BF16Frag a;
    a.u[0] = *(const uint4*)(arow + k0 + 8 * half);
    a.u[1] = *(const uint4*)(arow + k0 + 16 + 8 * half);
    BF16Frag bb;
    bb.u[0] = *(const uint4*)(brow + k0 + 8 * half);
    bb.u[1] = *(const uint4*)(brow + k0 + 16 + 8 * half);
    acc = __builtin_amdgcn_wmma_f32_16x16x32_bf16(false, a.v, false, bb.v,
                                                  (short)0, acc, false, false);
  }
#pragma unroll
  for (int r = 0; r < 8; ++r) {
    const int mr = mt * 16 + r + 8 * half;
    z[(size_t)mr * GDIM + nt * 16 + col] += acc[r];
  }
}

// ---------------- per-step pointwise: masters + gates + state update ----------------

__global__ void __launch_bounds__(256) pointwise_step(
    const float* __restrict__ z,        // [B][G] (z for this step, bias included)
    float* __restrict__ cstate,         // [B][U] f32
    unsigned short* __restrict__ hb,    // [B][U] bf16 (next step's GEMM input)
    float* __restrict__ out,            // [B][T][U] f32
    int t)
{
  __shared__ float fm[LVLS];
  __shared__ float im[LVLS];
  const int b = blockIdx.x;
  const float* zr = z + (size_t)b * GDIM;

  if (threadIdx.x == 0) {
    // left-to-right cumulative softmax of zr[0:16]
    float mx = zr[0];
    for (int i = 1; i < LVLS; ++i) mx = fmaxf(mx, zr[i]);
    float e[LVLS], s = 0.0f;
    for (int i = 0; i < LVLS; ++i) { e[i] = __expf(zr[i] - mx); s += e[i]; }
    const float inv = 1.0f / s;
    float cum = 0.0f;
    for (int i = 0; i < LVLS; ++i) { cum += e[i] * inv; fm[i] = cum; }
  } else if (threadIdx.x == 32) {
    // right-to-left cumulative softmax of zr[16:32]
    float mx = zr[LVLS];
    for (int i = 1; i < LVLS; ++i) mx = fmaxf(mx, zr[LVLS + i]);
    float e[LVLS], s = 0.0f;
    for (int i = 0; i < LVLS; ++i) { e[i] = __expf(zr[LVLS + i] - mx); s += e[i]; }
    const float inv = 1.0f / s;
    float cum = 0.0f;
    for (int i = LVLS - 1; i >= 0; --i) { cum += e[i] * inv; im[i] = cum; }
  }
  __syncthreads();

  for (int u = threadIdx.x; u < UDIM; u += blockDim.x) {
    const int l = u >> 6;
    const float fmv = fm[l];
    const float imv = im[l];
    const float fg = 1.0f / (1.0f + __expf(-zr[32 + u]));
    const float ig = 1.0f / (1.0f + __expf(-zr[32 + 1024 + u]));
    const float og = 1.0f / (1.0f + __expf(-zr[32 + 2048 + u]));
    const float ci = tanhf(zr[32 + 3072 + u]);
    const float cp = cstate[(size_t)b * UDIM + u];
    const float ov = fmv * imv;
    const float co = ov * (fg * cp + ig * ci) + (fmv - ov) * cp + (imv - ov) * ci;
    const float h  = og * tanhf(co);
    cstate[(size_t)b * UDIM + u] = co;
    hb[(size_t)b * UDIM + u] = f32_to_bf16_rne(h);
    out[((size_t)b * TDIM + t) * UDIM + u] = h;
  }
}

// ---------------- launcher ----------------

extern "C" void kernel_launch(void* const* d_in, const int* in_sizes, int n_in,
                              void* d_out, int out_size, void* d_ws, size_t ws_size,
                              hipStream_t stream) {
  const float* x       = (const float*)d_in[0];   // [B,T,D]
  const float* kernelW = (const float*)d_in[1];   // [D,G]
  const float* rkernel = (const float*)d_in[2];   // [U,G]
  const float* bias    = (const float*)d_in[3];   // [G]
  float* out = (float*)d_out;                     // [B,T,U]

  // workspace carve-out (~625 MB total; zx dominates at 541 MB)
  char* ws = (char*)d_ws;
  size_t off = 0;
  auto carve = [&](size_t bytes) -> void* {
    void* p = ws + off;
    off = (off + bytes + 255) & ~(size_t)255;
    return p;
  };
  float*          zx  = (float*)         carve((size_t)TDIM * BDIM * GDIM * 4);
  unsigned short* xb  = (unsigned short*)carve((size_t)BDIM * TDIM * DDIM * 2);
  unsigned short* wt  = (unsigned short*)carve((size_t)GDIM * DDIM * 2);
  unsigned short* rt  = (unsigned short*)carve((size_t)GDIM * UDIM * 2);
  unsigned short* hb  = (unsigned short*)carve((size_t)BDIM * UDIM * 2);
  float*          cst = (float*)         carve((size_t)BDIM * UDIM * 4);

  const int nx = BDIM * TDIM * DDIM;
  cvt_f32_to_bf16<<<(nx + 255) / 256, 256, 0, stream>>>(x, xb, nx);
  const int nw = DDIM * GDIM;
  transpose_to_bf16<<<(nw + 255) / 256, 256, 0, stream>>>(kernelW, wt, DDIM, GDIM);
  transpose_to_bf16<<<(nw + 255) / 256, 256, 0, stream>>>(rkernel, rt, UDIM, GDIM);
  const int ns = BDIM * UDIM;
  init_state<<<(ns + 255) / 256, 256, 0, stream>>>(cst, hb, ns);

  // big GEMM: 2048 M-tiles x 129 N-groups = 264192 waves, 8 waves/block
  {
    const long long waves = 2048LL * 129LL;
    const int blocks = (int)((waves + 7) / 8);
    gemm_zx<<<blocks, 256, 0, stream>>>(xb, wt, bias, zx);
  }

  // sequential scan: per step, recurrent GEMM (1032 waves) + pointwise (64 blocks)
  for (int t = 0; t < TDIM; ++t) {
    float* zslab = zx + (size_t)t * BDIM * GDIM;
    gemm_rec<<<129, 256, 0, stream>>>(hb, rt, zslab);
    pointwise_step<<<BDIM, 256, 0, stream>>>(zslab, cst, hb, out, t);
  }
}
